// ContrastiveLoss_3478923510424
// MI455X (gfx1250) — matrix operand order repaired and checked by reading, not compiled, not run on hardware
//
#include <hip/hip_runtime.h>
#include <math.h>

// ---------------------------------------------------------------------------
// ContrastiveLoss for MI455X (gfx1250), bf16x3 WMMA GEMM + fused log-sum-exp.
//   loss = sum_i [ log(sum_j exp(2 * <ne_i, ne_j>)) - 2 * <ne_i, nt_i> ] / (2N)
// Round 2: 2x2 register blocking per wave -> 12 WMMA / 16 loads per K-step,
// 4 independent accumulation chains, halved L2 B-traffic.
// ---------------------------------------------------------------------------

typedef __attribute__((ext_vector_type(16))) __bf16 v16bf;
typedef __attribute__((ext_vector_type(8)))  __bf16 bf16x8;
typedef __attribute__((ext_vector_type(4)))  __bf16 bf16x4;
typedef __attribute__((ext_vector_type(8)))  float  v8f;

#define DIM 512                 // feature dim (fixed by reference)
#define ROWS_PER_BLOCK 32       // output rows owned by one block (2 M-tiles)
#define COLS_PER_ITER 256       // 8 waves x 2 N-tiles x 16 cols per iteration

static __device__ __forceinline__ v16bf join8(bf16x8 a, bf16x8 b) {
    union { v16bf v; bf16x8 h[2]; } u;
    u.h[0] = a; u.h[1] = b;
    return u.v;
}

static __device__ __forceinline__ v8f wmma_bf16(v16bf a, v16bf b, v8f c) {
    return __builtin_amdgcn_wmma_f32_16x16x32_bf16(false, a, false, b,
                                                   (short)0, c, false, false);
}

// ---------------------------------------------------------------------------
// Kernel 1: normalize rows of embeddings; emit bf16 hi/lo split of ne and
// pos[i] = cos(emb_i, tgt_i).  One 128-thread block per row (4 floats/thread).
// ---------------------------------------------------------------------------
__global__ void __launch_bounds__(128) normalize_split_kernel(
    const float* __restrict__ emb, const float* __restrict__ tgt,
    __bf16* __restrict__ ne_hi, __bf16* __restrict__ ne_lo,
    float* __restrict__ pos) {
    const int row = blockIdx.x;
    const int t   = threadIdx.x;

    const float4* e4 = (const float4*)(emb + (size_t)row * DIM);
    const float4* g4 = (const float4*)(tgt + (size_t)row * DIM);
    float4 ev = e4[t];
    float4 gv = g4[t];

    float se = ev.x*ev.x + ev.y*ev.y + ev.z*ev.z + ev.w*ev.w;
    float st = gv.x*gv.x + gv.y*gv.y + gv.z*gv.z + gv.w*gv.w;
    float dp = ev.x*gv.x + ev.y*gv.y + ev.z*gv.z + ev.w*gv.w;

    #pragma unroll
    for (int m = 16; m >= 1; m >>= 1) {
        se += __shfl_xor(se, m, 32);
        st += __shfl_xor(st, m, 32);
        dp += __shfl_xor(dp, m, 32);
    }
    __shared__ float red[12];
    const int wid = t >> 5;
    if ((t & 31) == 0) { red[wid] = se; red[4 + wid] = st; red[8 + wid] = dp; }
    __syncthreads();
    se = red[0] + red[1] + red[2]  + red[3];
    st = red[4] + red[5] + red[6]  + red[7];
    dp = red[8] + red[9] + red[10] + red[11];

    const float inv_e = 1.0f / fmaxf(sqrtf(se), 1e-12f);
    const float inv_t = 1.0f / fmaxf(sqrtf(st), 1e-12f);
    if (t == 0) pos[row] = dp * inv_e * inv_t;

    float xs[4] = { ev.x * inv_e, ev.y * inv_e, ev.z * inv_e, ev.w * inv_e };
    bf16x4 h, l;
    #pragma unroll
    for (int i = 0; i < 4; ++i) {
        __bf16 hb = (__bf16)xs[i];
        h[i] = hb;
        l[i] = (__bf16)(xs[i] - (float)hb);   // residual for bf16x3 GEMM
    }
    *(bf16x4*)(ne_hi + (size_t)row * DIM + t * 4) = h;
    *(bf16x4*)(ne_lo + (size_t)row * DIM + t * 4) = l;
}

// ---------------------------------------------------------------------------
// Kernel 2: each block owns 32 rows; every wave computes BOTH 16-row M-tiles
// for its own 2 column tiles (cols j0 + wid*16 and j0 + wid*16 + 128).
// bf16x3 WMMA accumulates neg; exp(2*neg) summed per row across the N loop.
// ---------------------------------------------------------------------------
__global__ void __launch_bounds__(256) gemm_lse_kernel(
    const __bf16* __restrict__ ne_hi, const __bf16* __restrict__ ne_lo,
    const float*  __restrict__ pos,   float* __restrict__ rowloss, int N) {

    __shared__ __bf16 sA_hi[ROWS_PER_BLOCK * DIM];   // 32 KB
    __shared__ __bf16 sA_lo[ROWS_PER_BLOCK * DIM];   // 32 KB
    __shared__ float  srow[ROWS_PER_BLOCK];

    const int t    = threadIdx.x;
    const int row0 = blockIdx.x * ROWS_PER_BLOCK;

    // Stage this block's 32 A-rows (hi & lo) into LDS, 16B per transaction.
    for (int idx = t * 8; idx < ROWS_PER_BLOCK * DIM; idx += 256 * 8) {
        const int r = idx >> 9;            // idx / DIM
        const int c = idx & (DIM - 1);
        *(bf16x8*)&sA_hi[idx] = *(const bf16x8*)&ne_hi[(size_t)(row0 + r) * DIM + c];
        *(bf16x8*)&sA_lo[idx] = *(const bf16x8*)&ne_lo[(size_t)(row0 + r) * DIM + c];
    }
    if (t < ROWS_PER_BLOCK) srow[t] = 0.0f;
    __syncthreads();

    const int wid    = t >> 5;
    const int lane   = t & 31;
    const int lrow   = lane & 15;
    const int hi16   = lane >> 4;          // 0 or 1
    const int koffA  = hi16 ? 16 : 0;      // A frag: lanes 16-31 hold K 16..31
    const int koffB  = hi16 ? 16 : 0;      // B frag: lanes 16-31 hold K 16..31

    // A-fragment base pointers for both M-tiles (rows lrow and lrow+16).
    const __bf16* sa_hi0 = &sA_hi[(size_t)(lrow)      * DIM];
    const __bf16* sa_lo0 = &sA_lo[(size_t)(lrow)      * DIM];
    const __bf16* sa_hi1 = &sA_hi[(size_t)(16 + lrow) * DIM];
    const __bf16* sa_lo1 = &sA_lo[(size_t)(16 + lrow) * DIM];

    float racc0[8] = {0.f, 0.f, 0.f, 0.f, 0.f, 0.f, 0.f, 0.f};  // rows 0..15
    float racc1[8] = {0.f, 0.f, 0.f, 0.f, 0.f, 0.f, 0.f, 0.f};  // rows 16..31

    for (int j0 = 0; j0 < N; j0 += COLS_PER_ITER) {
        const int c0 = j0 + wid * 16;          // first column tile
        const int c1 = c0 + 128;               // second column tile
        const __bf16* bh0 = ne_hi + (size_t)(c0 + lrow) * DIM + koffB;
        const __bf16* bl0 = ne_lo + (size_t)(c0 + lrow) * DIM + koffB;
        const __bf16* bh1 = ne_hi + (size_t)(c1 + lrow) * DIM + koffB;
        const __bf16* bl1 = ne_lo + (size_t)(c1 + lrow) * DIM + koffB;

        // prefetch next iteration's B rows (global_prefetch_b8)
        if (j0 + COLS_PER_ITER < N) {
            __builtin_prefetch(ne_hi + (size_t)(c0 + COLS_PER_ITER + lrow) * DIM, 0, 1);
            __builtin_prefetch(ne_lo + (size_t)(c0 + COLS_PER_ITER + lrow) * DIM, 0, 1);
            __builtin_prefetch(ne_hi + (size_t)(c1 + COLS_PER_ITER + lrow) * DIM, 0, 1);
            __builtin_prefetch(ne_lo + (size_t)(c1 + COLS_PER_ITER + lrow) * DIM, 0, 1);
        }

        v8f acc00 = {0.f,0.f,0.f,0.f,0.f,0.f,0.f,0.f};   // M-tile 0 x col tile 0
        v8f acc01 = {0.f,0.f,0.f,0.f,0.f,0.f,0.f,0.f};   // M-tile 0 x col tile 1
        v8f acc10 = {0.f,0.f,0.f,0.f,0.f,0.f,0.f,0.f};   // M-tile 1 x col tile 0
        v8f acc11 = {0.f,0.f,0.f,0.f,0.f,0.f,0.f,0.f};   // M-tile 1 x col tile 1

        #pragma unroll 2
        for (int k0 = 0; k0 < DIM; k0 += 32) {
            // A fragments (LDS): two 16B chunks per lane, per M-tile, hi & lo.
            v16bf ah0 = join8(*(const bf16x8*)(sa_hi0 + k0 + koffA),
                              *(const bf16x8*)(sa_hi0 + k0 + koffA + 8));
            v16bf al0 = join8(*(const bf16x8*)(sa_lo0 + k0 + koffA),
                              *(const bf16x8*)(sa_lo0 + k0 + koffA + 8));
            v16bf ah1 = join8(*(const bf16x8*)(sa_hi1 + k0 + koffA),
                              *(const bf16x8*)(sa_hi1 + k0 + koffA + 8));
            v16bf al1 = join8(*(const bf16x8*)(sa_lo1 + k0 + koffA),
                              *(const bf16x8*)(sa_lo1 + k0 + koffA + 8));
            // B fragments (global, L2-resident): one 32B chunk per lane.
            v16bf bhv0 = *(const v16bf*)(bh0 + k0);
            v16bf blv0 = *(const v16bf*)(bl0 + k0);
            v16bf bhv1 = *(const v16bf*)(bh1 + k0);
            v16bf blv1 = *(const v16bf*)(bl1 + k0);

            // 12 WMMAs, round-robin across 4 independent accumulator chains
            // (dependent WMMAs are 4 issues apart).  bf16x3: hh + hl + lh.
            acc00 = wmma_bf16(ah0, bhv0, acc00);
            acc01 = wmma_bf16(ah0, bhv1, acc01);
            acc10 = wmma_bf16(ah1, bhv0, acc10);
            acc11 = wmma_bf16(ah1, bhv1, acc11);

            acc00 = wmma_bf16(ah0, blv0, acc00);
            acc01 = wmma_bf16(ah0, blv1, acc01);
            acc10 = wmma_bf16(ah1, blv0, acc10);
            acc11 = wmma_bf16(ah1, blv1, acc11);

            acc00 = wmma_bf16(al0, bhv0, acc00);
            acc01 = wmma_bf16(al0, bhv1, acc01);
            acc10 = wmma_bf16(al1, bhv0, acc10);
            acc11 = wmma_bf16(al1, bhv1, acc11);
        }

        // neg in [-1,1] -> exp(2*neg) <= e^2: no max-shift needed.
        #pragma unroll
        for (int v = 0; v < 8; ++v) {
            racc0[v] += __expf(2.0f * acc00[v]) + __expf(2.0f * acc01[v]);
            racc1[v] += __expf(2.0f * acc10[v]) + __expf(2.0f * acc11[v]);
        }
    }

    // C layout: lane holds N=lane%16; M = v (lanes 0-15) or v+8 (lanes 16-31).
    // Reduce each row over the 16 lanes holding its columns.
    #pragma unroll
    for (int v = 0; v < 8; ++v) {
        float x0 = racc0[v];
        float x1 = racc1[v];
        x0 += __shfl_xor(x0, 1, 32);  x1 += __shfl_xor(x1, 1, 32);
        x0 += __shfl_xor(x0, 2, 32);  x1 += __shfl_xor(x1, 2, 32);
        x0 += __shfl_xor(x0, 4, 32);  x1 += __shfl_xor(x1, 4, 32);
        x0 += __shfl_xor(x0, 8, 32);  x1 += __shfl_xor(x1, 8, 32);
        racc0[v] = x0;
        racc1[v] = x1;
    }
    if (lrow == 0) {
        const int mbase = hi16 ? 8 : 0;
        #pragma unroll
        for (int v = 0; v < 8; ++v) {
            atomicAdd(&srow[mbase + v],      racc0[v]);
            atomicAdd(&srow[16 + mbase + v], racc1[v]);
        }
    }
    __syncthreads();

    if (t < ROWS_PER_BLOCK) {
        // loss_partial = log(sum_j exp(2*neg)) - 2*pos
        rowloss[row0 + t] = logf(srow[t]) - 2.0f * pos[row0 + t];
    }
}

// ---------------------------------------------------------------------------
// Kernel 3: final reduction, out = sum(rowloss) / (2N)
// ---------------------------------------------------------------------------
__global__ void __launch_bounds__(256) reduce_loss_kernel(
    const float* __restrict__ rowloss, float* __restrict__ out, int N) {
    float s = 0.0f;
    for (int i = threadIdx.x; i < N; i += 256) s += rowloss[i];
    #pragma unroll
    for (int m = 16; m >= 1; m >>= 1) s += __shfl_xor(s, m, 32);
    __shared__ float red[8];
    if ((threadIdx.x & 31) == 0) red[threadIdx.x >> 5] = s;
    __syncthreads();
    if (threadIdx.x == 0) {
        float tot = 0.0f;
        #pragma unroll
        for (int w = 0; w < 8; ++w) tot += red[w];
        out[0] = tot / (2.0f * (float)N);
    }
}

// ---------------------------------------------------------------------------
extern "C" void kernel_launch(void* const* d_in, const int* in_sizes, int n_in,
                              void* d_out, int out_size, void* d_ws, size_t ws_size,
                              hipStream_t stream) {
    const float* emb = (const float*)d_in[0];
    const float* tgt = (const float*)d_in[1];
    const int N = in_sizes[0] / DIM;          // 8192

    // Workspace layout: ne_hi (N*DIM bf16) | ne_lo (N*DIM bf16) | pos | rowloss
    char* ws = (char*)d_ws;
    __bf16* ne_hi = (__bf16*)ws;
    __bf16* ne_lo = (__bf16*)(ws + (size_t)N * DIM * sizeof(__bf16));
    float*  pos   = (float*)(ws + (size_t)2 * N * DIM * sizeof(__bf16));
    float*  rowls = pos + N;

    normalize_split_kernel<<<N, 128, 0, stream>>>(emb, tgt, ne_hi, ne_lo, pos);
    gemm_lse_kernel<<<N / ROWS_PER_BLOCK, 256, 0, stream>>>(ne_hi, ne_lo, pos, rowls, N);
    reduce_loss_kernel<<<1, 256, 0, stream>>>(rowls, (float*)d_out, N);
}
